// EDRHSBuilder_83597243449885
// MI455X (gfx1250) — matrix operand order repaired and checked by reading, not compiled, not run on hardware
//
#include <hip/hip_runtime.h>
#include <hip/hip_bf16.h>

// ---------------- problem constants (match reference) ----------------
#define Bn 256
#define Tn 168
#define Gn 128
#define Sn 32
#define Kn 4
#define Pn 4
#define NZn 50000
#define NINEQ 249040
#define NEQ 27048

// per-row offsets inside h
#define H_PG_LB   0        // T*P = 672
#define H_PG_UB   672
#define H_ST_MAXL 1344     // T*S = 5376
#define H_ST_MINL 6720
#define H_ST_MAXC 12096
#define H_ST_MINC 17472
#define H_ST_MAXD 22848
#define H_ST_MIND 28224
#define H_ST_MAXE 33600    // S = 32
#define H_ST_MINE 33632
#define H_PA_UB   33664    // T*G = 21504
#define H_SEG_UB  55168    // T*G*K = 86016
#define H_Z_PA    141184
#define H_Z_SEG   162688
#define H_LOAD    248704   // T
#define H_Z_T     248872   // T

// per-row offsets inside b
#define B_INIT    0        // S
#define B_BAL     26880    // T

// global flat offsets in d_out
#define QOFF ((size_t)Bn * NZn)                       // 12,800,000
#define BOFF (QOFF + (size_t)Bn * NINEQ)              // 76,554,240

typedef float  f4v  __attribute__((ext_vector_type(4)));
typedef float  v8f  __attribute__((ext_vector_type(8)));
typedef _Float16 v16h __attribute__((ext_vector_type(16)));

__device__ __forceinline__ void nt_store(float* p, float v) {
    __builtin_nontemporal_store(v, p);
}
__device__ __forceinline__ void nt_store4(float* p, f4v v) {
    __builtin_nontemporal_store(v, (f4v*)p);
}

// ---------------- q: broadcast q_base over 256 rows ----------------
__global__ void q_fill_kernel(const float* __restrict__ q_base, float* __restrict__ out) {
    int i = blockIdx.x * 256 + threadIdx.x;            // float4 index within row
    if (i < NZn / 4) {
        f4v v = ((const f4v*)q_base)[i];
        f4v* dst = (f4v*)out + (size_t)blockIdx.y * (NZn / 4) + i;
        __builtin_nontemporal_store(v, dst);
    }
}

// ---------------- h: per (b,t) — pg bounds, load copy, z_t ----------------
__global__ void h_bt_kernel(const float* __restrict__ load,
                            const float* __restrict__ solar_max,
                            const float* __restrict__ wind_max,
                            const float* __restrict__ pg_min_power,
                            const float* __restrict__ pg_max_power,
                            float* __restrict__ out) {
    int idx = blockIdx.x * 256 + threadIdx.x;          // 0 .. B*T-1 (exact)
    int b = idx / Tn, t = idx % Tn;
    float* hrow = out + QOFF + (size_t)b * NINEQ;

    f4v pmin = *(const f4v*)pg_min_power;
    f4v lb = -pmin;
    f4v ub = *(const f4v*)pg_max_power;
    ub.x = solar_max[idx];                              // IDX_SOLAR = 0
    ub.y = wind_max[idx];                               // IDX_WIND  = 1

    nt_store4(hrow + H_PG_LB + t * Pn, lb);
    nt_store4(hrow + H_PG_UB + t * Pn, ub);
    nt_store(hrow + H_LOAD + t, load[idx]);
    nt_store(hrow + H_Z_T + t, 0.0f);
}

// ---------------- h: per (b,t,s) — storage bounds (+end rows at t==0) ----------------
__global__ void h_st_kernel(const float* __restrict__ is_charging,
                            const float* __restrict__ is_discharging,
                            const float* __restrict__ st_max_levels,
                            const float* __restrict__ st_min_levels,
                            const float* __restrict__ st_max_charge,
                            const float* __restrict__ st_min_charge,
                            const float* __restrict__ st_max_discharge,
                            const float* __restrict__ st_min_discharge,
                            const float* __restrict__ st_max_end,
                            const float* __restrict__ st_min_end,
                            float* __restrict__ out) {
    int idx = blockIdx.x * 256 + threadIdx.x;          // 0 .. B*T*S-1 (exact)
    int s = idx % Sn;
    int t = (idx / Sn) % Tn;
    int b = idx / (Sn * Tn);
    float* hrow = out + QOFF + (size_t)b * NINEQ;
    int col = t * Sn + s;

    float ch = is_charging[idx];
    float dis = is_discharging[idx];

    nt_store(hrow + H_ST_MAXL + col,  st_max_levels[s]);
    nt_store(hrow + H_ST_MINL + col, -st_min_levels[s]);
    nt_store(hrow + H_ST_MAXC + col,  st_max_charge[s] * ch);
    nt_store(hrow + H_ST_MINC + col, -st_min_charge[s] * ch);
    nt_store(hrow + H_ST_MAXD + col,  st_max_discharge[s] * dis);
    nt_store(hrow + H_ST_MIND + col, -st_min_discharge[s] * dis);
    if (t == 0) {
        nt_store(hrow + H_ST_MAXE + s,  st_max_end[s]);
        nt_store(hrow + H_ST_MINE + s, -st_min_end[s]);
    }
}

// ---------------- h: per (b,t,g) — thermal bounds + zero blocks ----------------
__global__ void h_pa_seg_kernel(const float* __restrict__ is_on,
                                const float* __restrict__ th_power_diff,
                                const float* __restrict__ th_seg_mw,
                                float* __restrict__ out) {
    int idx = blockIdx.x * 256 + threadIdx.x;          // 0 .. B*T*G-1 (exact)
    int g = idx % Gn;
    int t = (idx / Gn) % Tn;
    int b = idx / (Gn * Tn);
    float* hrow = out + QOFF + (size_t)b * NINEQ;
    int col = t * Gn + g;

    float on = is_on[idx];
    f4v seg = ((const f4v*)th_seg_mw)[g];              // th_seg_mw is (G, 4)
    seg *= on;
    f4v z4 = {0.0f, 0.0f, 0.0f, 0.0f};

    nt_store(hrow + H_PA_UB + col, th_power_diff[g] * on);
    nt_store4(hrow + H_SEG_UB + col * Kn, seg);
    nt_store(hrow + H_Z_PA + col, 0.0f);
    nt_store4(hrow + H_Z_SEG + col * Kn, z4);
}

// ---------------- b: st_init + zeros for cols [0, 26880) ----------------
__global__ void b_fill_kernel(const float* __restrict__ st_init_levels,
                              float* __restrict__ out) {
    int idx = blockIdx.x * 256 + threadIdx.x;          // 0 .. B*6720-1 (exact)
    int i4 = idx % (B_BAL / 4);
    int b = idx / (B_BAL / 4);
    int c = i4 * 4;
    float* brow = out + BOFF + (size_t)b * NEQ;
    f4v v = {0.0f, 0.0f, 0.0f, 0.0f};
    if (c < Sn) v = *(const f4v*)(st_init_levels + c);
    nt_store4(brow + c, v);
}

// ---------------- b: power balance via WMMA f16 (hi/lo exact split) ----------------
// pmin[r] = sum_g is_on[r,g] * th_min_power[g], r = b*T + t (43008 rows, K=128)
// One wave computes a 16-row tile with v_wmma_f32_16x16x32_f16; th is split into
// f16 hi + f16 lo so the f32 accumulation is exact to ~2^-22 relative.
__global__ void pmin_wmma_kernel(const float* __restrict__ is_on,
                                 const float* __restrict__ th_min_power,
                                 const float* __restrict__ load,
                                 float* __restrict__ out) {
    const int lane = threadIdx.x & 31;
    const int wave = threadIdx.x >> 5;
    const int gwave = blockIdx.x * 8 + wave;           // blockDim = 256 -> 8 waves
    const int row0 = gwave * 16;                       // rows: 0 .. 43007 (exact)
    const int m = lane & 15;
    const int half = lane >> 4;
    const int row = row0 + m;

    v8f acc = {0.f, 0.f, 0.f, 0.f, 0.f, 0.f, 0.f, 0.f};

    #pragma unroll
    for (int kk = 0; kk < 4; ++kk) {
        const int k0 = kk * 32;
        // A tile (16x32 f16): lane holds M=row; VGPR j: K = 8*half + 2j (+16 for j>=4)
        const float* arow = is_on + (size_t)row * Gn + k0 + 8 * half;
        v16h a;
        #pragma unroll
        for (int j = 0; j < 8; ++j) {
            int kofs = (j < 4) ? (2 * j) : (16 + 2 * (j - 4));
            a[2 * j]     = (_Float16)arow[kofs];
            a[2 * j + 1] = (_Float16)arow[kofs + 1];
        }
        // B tile (32x16 f16), th replicated across all 16 columns:
        // lane half h holds K = k0 + 8*h + [0..7], then k0 + 16 + 8*h + [0..7]
        const float* brow = th_min_power + k0 + 8 * half;
        v16h bh, bl;
        #pragma unroll
        for (int j = 0; j < 16; ++j) {
            float v = brow[(j < 8) ? j : (j + 8)];
            _Float16 hi = (_Float16)v;
            bh[j] = hi;
            bl[j] = (_Float16)(v - (float)hi);
        }
        acc = __builtin_amdgcn_wmma_f32_16x16x32_f16(false, a, false, bh,
                                                     (short)0, acc, false, false);
        acc = __builtin_amdgcn_wmma_f32_16x16x32_f16(false, a, false, bl,
                                                     (short)0, acc, false, false);
    }

    // D[m][n] identical across n. acc[r] on half h = row row0 + 8*h + r.
    if ((lane & 15) == 0) {
        #pragma unroll
        for (int r = 0; r < 8; ++r) {
            int rr = row0 + half * 8 + r;
            int b = rr / Tn;
            int t = rr % Tn;
            nt_store(out + BOFF + (size_t)b * NEQ + B_BAL + t, load[rr] - acc[r]);
        }
    }
}

// ---------------- launcher ----------------
extern "C" void kernel_launch(void* const* d_in, const int* in_sizes, int n_in,
                              void* d_out, int out_size, void* d_ws, size_t ws_size,
                              hipStream_t stream) {
    const float* load            = (const float*)d_in[0];
    const float* solar_max       = (const float*)d_in[1];
    const float* wind_max        = (const float*)d_in[2];
    const float* is_on           = (const float*)d_in[3];
    const float* is_charging     = (const float*)d_in[4];
    const float* is_discharging  = (const float*)d_in[5];
    const float* pg_min_power    = (const float*)d_in[6];
    const float* pg_max_power    = (const float*)d_in[7];
    const float* st_max_levels   = (const float*)d_in[8];
    const float* st_min_levels   = (const float*)d_in[9];
    const float* st_max_charge   = (const float*)d_in[10];
    const float* st_min_charge   = (const float*)d_in[11];
    const float* st_max_discharge= (const float*)d_in[12];
    const float* st_min_discharge= (const float*)d_in[13];
    const float* st_max_end      = (const float*)d_in[14];
    const float* st_min_end      = (const float*)d_in[15];
    const float* st_init_levels  = (const float*)d_in[16];
    const float* th_power_diff   = (const float*)d_in[17];
    const float* th_seg_mw       = (const float*)d_in[18];
    const float* th_min_power    = (const float*)d_in[19];
    const float* q_base          = (const float*)d_in[20];
    float* out = (float*)d_out;

    // q: 256 rows x 12500 float4
    dim3 gq((NZn / 4 + 255) / 256, Bn);
    q_fill_kernel<<<gq, 256, 0, stream>>>(q_base, out);

    // h blocks
    h_bt_kernel<<<(Bn * Tn) / 256, 256, 0, stream>>>(
        load, solar_max, wind_max, pg_min_power, pg_max_power, out);
    h_st_kernel<<<(Bn * Tn * Sn) / 256, 256, 0, stream>>>(
        is_charging, is_discharging, st_max_levels, st_min_levels,
        st_max_charge, st_min_charge, st_max_discharge, st_min_discharge,
        st_max_end, st_min_end, out);
    h_pa_seg_kernel<<<(Bn * Tn * Gn) / 256, 256, 0, stream>>>(
        is_on, th_power_diff, th_seg_mw, out);

    // b blocks
    b_fill_kernel<<<(Bn * (B_BAL / 4)) / 256, 256, 0, stream>>>(st_init_levels, out);
    pmin_wmma_kernel<<<(Bn * Tn) / (16 * 8), 256, 0, stream>>>(
        is_on, th_min_power, load, out);
}
